// NeuralODE_22763326669447
// MI455X (gfx1250) — matrix operand order repaired and checked by reading, not compiled, not run on hardware
//
#include <hip/hip_runtime.h>

typedef __attribute__((ext_vector_type(16))) _Float16 v16h;
typedef __attribute__((ext_vector_type(8)))  float    v8f;
typedef __attribute__((ext_vector_type(4)))  float    v4f;

namespace {
constexpr int   kT = 40;
constexpr float cA21 = 0.161f;
constexpr float cA31 = -0.008480655492356989f, cA32 = 0.335480655492357f;
constexpr float cA41 = 2.8971530571054935f,  cA42 = -6.359448489975075f,  cA43 = 4.3622954328695815f;
constexpr float cA51 = 5.325864828439257f,   cA52 = -11.748883564062828f, cA53 = 7.4955393428898365f, cA54 = -0.09249506636175525f;
constexpr float cA61 = 5.86145544294642f,    cA62 = -12.92096931784711f,  cA63 = 8.159367898576159f,  cA64 = -0.071584973281401f, cA65 = -0.028269050394068383f;
constexpr float cB1 = 0.09646076681806523f, cB2 = 0.01f, cB3 = 0.4798896504144996f;
constexpr float cB4 = 1.379008574103742f,   cB5 = -3.290069515436081f, cB6 = 2.324710524099774f;
constexpr float LOG2E = 1.4426950408889634f;
constexpr float LN2   = 0.6931471805599453f;
}

__device__ __forceinline__ v8f wmma16(v16h a, v16h b, v8f c) {
  // D = A(16x32 f16) * B(32x16 f16) + C(16x16 f32)
  return __builtin_amdgcn_wmma_f32_16x16x32_f16(false, a, false, b, (short)0, c, false, false);
}

__global__ __launch_bounds__(256, 1)
void node_tsit5_kernel(const float* __restrict__ x0s,
                       const float* __restrict__ t_eval,
                       const float* __restrict__ W0, const float* __restrict__ b0,
                       const float* __restrict__ W1, const float* __restrict__ b1,
                       const float* __restrict__ W2, const float* __restrict__ b2,
                       const float* __restrict__ W3, const float* __restrict__ b3,
                       float* __restrict__ out)
{
  const int lane = threadIdx.x & 31;
  const int wave = threadIdx.x >> 5;
  const int tile = blockIdx.x * (blockDim.x >> 5) + wave;   // 512 tiles of 16 trajectories
  const int hi   = lane >> 4;                               // half-wave select
  const int col  = lane & 15;                               // batch col / feature row
  const int bidx = tile * 16 + col;                         // trajectory index

  // ---- Weights as WMMA A operands (W^T) in table K order:
  // slot j of half hi holds W row  j + (j&8) + hi*8,  matching the B pack below
  // (slots 0..7 = features 8*hi+j, slots 8..15 = features 16+8*hi+j).
  auto loadA = [&](const float* W, int kin, int nout, int m) {
    v16h r;
#pragma unroll
    for (int j = 0; j < 16; ++j) {
      int f = j + (j & 8) + hi * 8;                          // input-feature row of W
      float v = (f < kin) ? W[f * nout + (m * 16 + col)] : 0.0f;  // W is [fan_in][fan_out]
      r[j] = (_Float16)v;
    }
    return r;
  };
  auto loadC = [&](const float* bb, int m) {                  // bias as C seed (D layout)
    v8f r;
#pragma unroll
    for (int j = 0; j < 8; ++j) r[j] = bb[m * 16 + hi * 8 + j];
    return r;
  };

  const v16h wa00 = loadA(W0, 16, 32, 0), wa01 = loadA(W0, 16, 32, 1);
  const v16h wa10 = loadA(W1, 32, 32, 0), wa11 = loadA(W1, 32, 32, 1);
  const v16h wa20 = loadA(W2, 32, 32, 0), wa21 = loadA(W2, 32, 32, 1);
  const v16h wa30 = loadA(W3, 32, 16, 0);
  const v8f  bc00 = loadC(b0, 0), bc01 = loadC(b0, 1);
  const v8f  bc10 = loadC(b1, 0), bc11 = loadC(b1, 1);
  const v8f  bc20 = loadC(b2, 0), bc21 = loadC(b2, 1);
  const v8f  bc30 = loadC(b3, 0);

  // ---- B-operand build: thanks to the K permutation, every lane packs its own
  // D-tile registers: slots 0..7 = tile0 vgprs, slots 8..15 = tile1 vgprs.
  auto buildB = [&](v8f a0, v8f a1) {
    v16h r;
#pragma unroll
    for (int j = 0; j < 8; ++j) {
      r[j]     = (_Float16)a0[j];
      r[8 + j] = (_Float16)a1[j];
    }
    return r;
  };
  auto buildB0 = [&](v8f a0) {                               // 16-feature input, upper K zero
    v16h r;
#pragma unroll
    for (int j = 0; j < 8; ++j) {
      r[j]     = (_Float16)a0[j];
      r[8 + j] = (_Float16)0.0f;
    }
    return r;
  };
  // softplus(x) = relu(x) + ln2*log2(1 + 2^(-|x*log2e|)).
  //  * |x|*log2e == |x*log2e| (log2e>0), so the neg+abs folds into v_exp_f32's
  //    source modifiers and the scale stays a plain (pk-)mul.
  //  * relu via v_fmed3_f32(x, 0, +inf): one op, no canonicalize.
  //  * exp2 arg <= 0 -> result in (0,1], 1+e in (1,2]: raw v_exp/v_log are safe.
  auto sp8 = [&](v8f v) {
    v8f r;
#pragma unroll
    for (int j = 0; j < 8; ++j) {
      float x = v[j];
      float m = x * LOG2E;
      float e = __builtin_amdgcn_exp2f(-fabsf(m));
      float relu = __builtin_amdgcn_fmed3f(x, 0.0f, __builtin_inff());
      r[j] = fmaf(LN2, __builtin_amdgcn_logf(1.0f + e), relu);
    }
    return r;
  };

  auto mlp = [&](v8f in) {
    v16h Bv = buildB0(in);
    v8f h0 = wmma16(wa00, Bv, bc00);
    v8f h1 = wmma16(wa01, Bv, bc01);
    h0 = sp8(h0); h1 = sp8(h1);
    Bv = buildB(h0, h1);
    v8f g0 = wmma16(wa10, Bv, bc10);
    v8f g1 = wmma16(wa11, Bv, bc11);
    g0 = sp8(g0); g1 = sp8(g1);
    Bv = buildB(g0, g1);
    h0 = wmma16(wa20, Bv, bc20);
    h1 = wmma16(wa21, Bv, bc21);
    h0 = sp8(h0); h1 = sp8(h1);
    Bv = buildB(h0, h1);
    return wmma16(wa30, Bv, bc30);   // 16 output features, single M tile
  };

  // ---- Load initial state (D layout matches contiguous memory slices).
  v8f y;
  {
    const v4f* p = (const v4f*)(x0s + (size_t)bidx * 16 + hi * 8);
    v4f u0 = p[0], u1 = p[1];
#pragma unroll
    for (int j = 0; j < 4; ++j) { y[j] = u0[j]; y[4 + j] = u1[j]; }
  }
  auto save = [&](int t) {
    v4f u0, u1;
#pragma unroll
    for (int j = 0; j < 4; ++j) { u0[j] = y[j]; u1[j] = y[4 + j]; }
    v4f* p = (v4f*)(out + ((size_t)bidx * kT + t) * 16 + hi * 8);
    p[0] = u0; p[1] = u1;
  };
  save(0);

#pragma unroll 1
  for (int t = 0; t < kT - 1; ++t) {
    const float dt = (t_eval[t + 1] - t_eval[t]) * 0.125f;
#pragma unroll 1
    for (int s = 0; s < 8; ++s) {
      v8f k1 = mlp(y);
      v8f yt;
#pragma unroll
      for (int j = 0; j < 8; ++j) yt[j] = fmaf(dt * cA21, k1[j], y[j]);
      v8f k2 = mlp(yt);
#pragma unroll
      for (int j = 0; j < 8; ++j) yt[j] = y[j] + dt * (cA31 * k1[j] + cA32 * k2[j]);
      v8f k3 = mlp(yt);
#pragma unroll
      for (int j = 0; j < 8; ++j) yt[j] = y[j] + dt * (cA41 * k1[j] + cA42 * k2[j] + cA43 * k3[j]);
      v8f k4 = mlp(yt);
#pragma unroll
      for (int j = 0; j < 8; ++j) yt[j] = y[j] + dt * (cA51 * k1[j] + cA52 * k2[j] + cA53 * k3[j] + cA54 * k4[j]);
      v8f k5 = mlp(yt);
#pragma unroll
      for (int j = 0; j < 8; ++j) yt[j] = y[j] + dt * (cA61 * k1[j] + cA62 * k2[j] + cA63 * k3[j] + cA64 * k4[j] + cA65 * k5[j]);
      v8f k6 = mlp(yt);
#pragma unroll
      for (int j = 0; j < 8; ++j)
        y[j] = y[j] + dt * (cB1 * k1[j] + cB2 * k2[j] + cB3 * k3[j] + cB4 * k4[j] + cB5 * k5[j] + cB6 * k6[j]);
    }
    save(t + 1);
  }
}

extern "C" void kernel_launch(void* const* d_in, const int* in_sizes, int n_in,
                              void* d_out, int out_size, void* d_ws, size_t ws_size,
                              hipStream_t stream) {
  const float* x0s = (const float*)d_in[0];
  const float* te  = (const float*)d_in[1];
  const float* W0  = (const float*)d_in[2];
  const float* b0  = (const float*)d_in[3];
  const float* W1  = (const float*)d_in[4];
  const float* b1  = (const float*)d_in[5];
  const float* W2  = (const float*)d_in[6];
  const float* b2  = (const float*)d_in[7];
  const float* W3  = (const float*)d_in[8];
  const float* b3  = (const float*)d_in[9];
  float* out = (float*)d_out;

  const int waves_per_block = 8;                 // 256 threads
  const int tiles = 8192 / 16;                   // 512 independent 16-trajectory tiles
  dim3 grid(tiles / waves_per_block), block(waves_per_block * 32);
  node_tsit5_kernel<<<grid, block, 0, stream>>>(x0s, te, W0, b0, W1, b1, W2, b2, W3, b3, out);
}
